// CrossAttFuseMod_37615323578957
// MI455X (gfx1250) — matrix operand order repaired and checked by reading, not compiled, not run on hardware
//
#include <hip/hip_runtime.h>
#include <cstdint>
#include <cstddef>

// ---------------------------------------------------------------------------
// CDNA5 (gfx1250) implementation of CrossAttFuseMod.
//
// Exact simplification: mixer cross-attention has sequence length 1 => softmax
// over the single score is identically 1.0, so _cross(x,y) = (y@Wv^T)@Wo^T+bo
// (Wq, Wk, attention_mask are mathematically dead).
//
// Precision: all GEMM operands bf16 (producers emit bf16 shadows), WMMA f32
// accumulate (v_wmma_f32_16x16x32_bf16). Softmax/LN/heads in f32.
// Data movement: double-buffered LDS staged with GLOBAL_LOAD_ASYNC_TO_LDS_B128
// (ASYNCcnt), one barrier per 64-wide K step, async loads overlap WMMA.
// ---------------------------------------------------------------------------

typedef __bf16 bf16;
typedef __bf16 v16bf __attribute__((ext_vector_type(16)));
typedef float  v8f   __attribute__((ext_vector_type(8)));
typedef int    v4i   __attribute__((ext_vector_type(4)));

#define AS1 __attribute__((address_space(1)))
#define AS3 __attribute__((address_space(3)))

#define LDS_P 72   // LDS tile row pitch in bf16 (64 K + 8 pad; keeps 16B align)

// ---- async global->LDS 16B copy (per-lane), tracked by ASYNCcnt ----
__device__ __forceinline__ void async_copy_b128(const bf16* g, bf16* l) {
#if __has_builtin(__builtin_amdgcn_global_load_async_to_lds_b128)
  __builtin_amdgcn_global_load_async_to_lds_b128(
      (AS1 v4i*)(void*)g, (AS3 v4i*)(void*)l, 0, 0);
#else
  const unsigned lds = (unsigned)(uintptr_t)(AS3 void*)(void*)l;
  asm volatile("global_load_async_to_lds_b128 %0, %1, off"
               :: "v"(lds), "v"((unsigned long long)(uintptr_t)g)
               : "memory");
#endif
}

__device__ __forceinline__ void wait_async0() {
#if __has_builtin(__builtin_amdgcn_s_wait_asynccnt)
  __builtin_amdgcn_s_wait_asynccnt(0);
#else
  asm volatile("s_wait_asynccnt 0" ::: "memory");
#endif
}

// ---------------- f32 -> bf16 conversion (grid-stride) ----------------
__global__ __launch_bounds__(256) void k_f32_to_bf16(const float* __restrict__ s,
                                                     bf16* __restrict__ d, int n) {
  int i = blockIdx.x * 256 + threadIdx.x;
  const int stride = gridDim.x * 256;
  for (; i < n; i += stride) d[i] = (bf16)s[i];
}

// ---------------- WMMA GEMM ----------------
// C(MxN,f32) [+ optional bf16 shadow Cb] = act( A(MxK,bf16) * op(B) + bias )
//   TRANS_B=false : B stored (N x K) row-major (weight layout), op(B)=B^T
//   TRANS_B=true  : B stored (K x N) row-major, op(B)=B (LDS transpose)
// Block: 256 threads = 8 wave32. Tile 128(M) x 128(N), K stepped by 64.
// Wave grid 2(M) x 4(N); each wave owns 64x32 = 4x2 WMMA accumulators.
// A/B tiles: double-buffered LDS, filled by async global->LDS b128 copies
// (B transposed synchronously in the TRANS_B case).
template<bool TRANS_B, bool RELU>
__global__ __launch_bounds__(256) void k_gemm_wmma(
    const bf16* __restrict__ A, int lda,
    const bf16* __restrict__ B, int ldb,
    const float* __restrict__ bias,
    float* __restrict__ C, bf16* __restrict__ Cb, int ldc,
    int K)
{
  __shared__ bf16 ldsA[2][128 * LDS_P];   // 2 x 18 KB
  __shared__ bf16 ldsB[2][128 * LDS_P];   // 2 x 18 KB
  const int t    = threadIdx.x;
  const int lane = t & 31;
  const int wave = t >> 5;
  const int m0   = blockIdx.y * 128;
  const int n0   = blockIdx.x * 128;
  const int wm   = (wave & 1) * 64;       // wave row offset in tile
  const int wn   = (wave >> 1) * 32;      // wave col offset in tile
  const int half = lane >> 4;             // 0: lanes 0-15, 1: lanes 16-31
  const int l15  = lane & 15;

  v8f acc[4][2] = {};

  // stage A tile (128 rows x 64 k) via async copies: 1024 16B chunks
  auto stageA = [&](int buf, int k0) {
#pragma unroll
    for (int i = 0; i < 4; ++i) {
      const int c   = t + i * 256;
      const int row = c >> 3;
      const int ko  = (c & 7) * 8;
      async_copy_b128(A + (size_t)(m0 + row) * lda + k0 + ko,
                      &ldsA[buf][row * LDS_P + ko]);
    }
  };
  // stage B tile as [n][k] (128 x 64)
  auto stageB = [&](int buf, int k0) {
    if (!TRANS_B) {
#pragma unroll
      for (int i = 0; i < 4; ++i) {
        const int c   = t + i * 256;
        const int row = c >> 3;
        const int ko  = (c & 7) * 8;
        async_copy_b128(B + (size_t)(n0 + row) * ldb + k0 + ko,
                        &ldsB[buf][row * LDS_P + ko]);
      }
    } else {
#pragma unroll
      for (int i = 0; i < 4; ++i) {
        const int c  = t + i * 256;       // 64(k) x 16 chunks of 8 n
        const int k  = c >> 4;            // 0..63
        const int nq = (c & 15) * 8;      // 0..120
        const uint4 v = *(const uint4*)(B + (size_t)(k0 + k) * ldb + n0 + nq);
        const bf16* e = (const bf16*)&v;
#pragma unroll
        for (int j = 0; j < 8; ++j) ldsB[buf][(nq + j) * LDS_P + k] = e[j];
      }
    }
  };

  const int nsteps = K >> 6;
  stageA(0, 0);
  stageB(0, 0);

  for (int s = 0; s < nsteps; ++s) {
    const int buf = s & 1;
    wait_async0();       // own async copies for stage s complete
    __syncthreads();     // everyone's stage-s data in LDS; prior reads done
    if (s + 1 < nsteps) {          // overlap next stage with compute
      stageA(buf ^ 1, (s + 1) << 6);
      stageB(buf ^ 1, (s + 1) << 6);
    }
#pragma unroll
    for (int ks = 0; ks < 64; ks += 32) {
      v16bf aF[4], bF[2];
#pragma unroll
      for (int mi = 0; mi < 4; ++mi) {
        // A 16x32 layout: lane<16 -> K {0..7,16..23}, lane>=16 -> {8..15,24..31}
        const bf16* p = &ldsA[buf][(wm + mi * 16 + l15) * LDS_P + ks + half * 8];
        uint4* fp = (uint4*)&aF[mi];
        fp[0] = *(const uint4*)p;
        fp[1] = *(const uint4*)(p + 16);
      }
#pragma unroll
      for (int nj = 0; nj < 2; ++nj) {
        // B 32x16 layout: lane holds column N=l15, lane<16 -> K {0..7,16..23}, lane>=16 -> {8..15,24..31}
        const bf16* p = &ldsB[buf][(wn + nj * 16 + l15) * LDS_P + ks + half * 8];
        uint4* fp = (uint4*)&bF[nj];
        fp[0] = *(const uint4*)p;
        fp[1] = *(const uint4*)(p + 16);
      }
#pragma unroll
      for (int mi = 0; mi < 4; ++mi)
#pragma unroll
        for (int nj = 0; nj < 2; ++nj)
          acc[mi][nj] = __builtin_amdgcn_wmma_f32_16x16x32_bf16(
              false, aF[mi], false, bF[nj], (short)0, acc[mi][nj], false, false);
    }
  }

  // ---- epilogue: + bias, optional ReLU, f32 C + optional bf16 shadow ----
  const bool hasShadow = (Cb != nullptr);
#pragma unroll
  for (int nj = 0; nj < 2; ++nj) {
    const int n  = n0 + wn + nj * 16 + l15;
    const float bv = bias ? bias[n] : 0.0f;
#pragma unroll
    for (int mi = 0; mi < 4; ++mi) {
      const int mb = m0 + wm + mi * 16 + half * 8;
#pragma unroll
      for (int v = 0; v < 8; ++v) {
        float val = acc[mi][nj][v] + bv;
        if (RELU) val = fmaxf(val, 0.0f);
        const size_t idx = (size_t)(mb + v) * ldc + n;
        C[idx] = val;
        if (hasShadow) Cb[idx] = (bf16)val;
      }
    }
  }
}

// ---------------- row-wise softmax (in place) + bf16 shadow ----------------
__global__ __launch_bounds__(256) void k_softmax_rows(float* __restrict__ S,
                                                      bf16* __restrict__ Sb,
                                                      int cols, float scale) {
  float* p = S + (size_t)blockIdx.x * cols;
  bf16*  q = Sb + (size_t)blockIdx.x * cols;
  __shared__ float red[256];
  const int t = threadIdx.x;
  float m = -3.4e38f;
  for (int c = t; c < cols; c += 256) m = fmaxf(m, p[c] * scale);
  red[t] = m; __syncthreads();
  for (int s = 128; s > 0; s >>= 1) { if (t < s) red[t] = fmaxf(red[t], red[t + s]); __syncthreads(); }
  m = red[0]; __syncthreads();
  float sum = 0.0f;
  for (int c = t; c < cols; c += 256) { float e = __expf(p[c] * scale - m); p[c] = e; sum += e; }
  red[t] = sum; __syncthreads();
  for (int s = 128; s > 0; s >>= 1) { if (t < s) red[t] += red[t + s]; __syncthreads(); }
  const float inv = 1.0f / red[0];
  for (int c = t; c < cols; c += 256) {
    const float v = p[c] * inv;
    p[c] = v;
    q[c] = (bf16)v;
  }
}

// ---------------- row-wise log-softmax (in place) ----------------
__global__ __launch_bounds__(256) void k_log_softmax_rows(float* __restrict__ S, int cols) {
  float* p = S + (size_t)blockIdx.x * cols;
  __shared__ float red[256];
  const int t = threadIdx.x;
  float m = -3.4e38f;
  for (int c = t; c < cols; c += 256) m = fmaxf(m, p[c]);
  red[t] = m; __syncthreads();
  for (int s = 128; s > 0; s >>= 1) { if (t < s) red[t] = fmaxf(red[t], red[t + s]); __syncthreads(); }
  m = red[0]; __syncthreads();
  float sum = 0.0f;
  for (int c = t; c < cols; c += 256) sum += __expf(p[c] - m);
  red[t] = sum; __syncthreads();
  for (int s = 128; s > 0; s >>= 1) { if (t < s) red[t] += red[t + s]; __syncthreads(); }
  const float ls = logf(red[0]);
  for (int c = t; c < cols; c += 256) p[c] = p[c] - m - ls;
}

// -------- residual + layernorm: O = LN(X+H)*g+b, + bf16 shadow --------
__global__ __launch_bounds__(256) void k_add_ln(const float* __restrict__ X,
                                                const float* __restrict__ H,
                                                const float* __restrict__ gam,
                                                const float* __restrict__ bet,
                                                float* __restrict__ O,
                                                bf16* __restrict__ Ob, int cols) {
  const size_t r = (size_t)blockIdx.x * cols;
  __shared__ float r0[256], r1[256];
  const int t = threadIdx.x;
  float s = 0.0f, sq = 0.0f;
  for (int c = t; c < cols; c += 256) {
    const float v = X[r + c] + H[r + c];
    O[r + c] = v; s += v; sq += v * v;
  }
  r0[t] = s; r1[t] = sq; __syncthreads();
  for (int st = 128; st > 0; st >>= 1) {
    if (t < st) { r0[t] += r0[t + st]; r1[t] += r1[t + st]; }
    __syncthreads();
  }
  const float mean = r0[0] / cols;
  const float var  = r1[0] / cols - mean * mean;
  const float rstd = rsqrtf(var + 1e-5f);
  for (int c = t; c < cols; c += 256) {
    const float v = (O[r + c] - mean) * rstd * gam[c] + bet[c];
    O[r + c]  = v;
    Ob[r + c] = (bf16)v;
  }
}

// -------- ali head: logits (N=2) + log-softmax, one block per row --------
__global__ __launch_bounds__(256) void k_ali(const float* __restrict__ X,
                                             const float* __restrict__ W,
                                             const float* __restrict__ b,
                                             float* __restrict__ out) {
  const size_t r = (size_t)blockIdx.x * 2048;
  __shared__ float r0[256], r1[256];
  const int t = threadIdx.x;
  float d0 = 0.0f, d1 = 0.0f;
  for (int c = t; c < 2048; c += 256) {
    const float v = X[r + c];
    d0 += v * W[c];
    d1 += v * W[2048 + c];
  }
  r0[t] = d0; r1[t] = d1; __syncthreads();
  for (int s = 128; s > 0; s >>= 1) {
    if (t < s) { r0[t] += r0[t + s]; r1[t] += r1[t + s]; }
    __syncthreads();
  }
  if (t == 0) {
    const float l0 = r0[0] + b[0], l1 = r1[0] + b[1];
    const float m  = fmaxf(l0, l1);
    const float ls = logf(__expf(l0 - m) + __expf(l1 - m));
    out[(size_t)blockIdx.x * 2 + 0] = l0 - m - ls;
    out[(size_t)blockIdx.x * 2 + 1] = l1 - m - ls;
  }
}

// ---------------------------------------------------------------------------
// Host orchestration
// ---------------------------------------------------------------------------
extern "C" void kernel_launch(void* const* d_in, const int* in_sizes, int n_in,
                              void* d_out, int out_size, void* d_ws, size_t ws_size,
                              hipStream_t stream) {
  (void)in_sizes; (void)n_in; (void)out_size; (void)ws_size;

  // Input index map (setup_inputs dict insertion order, recursively flattened):
  // 0 text, 1 graph, 2 attention_mask (dead: seq-len-1 softmax == 1),
  // 3 te1_W, 4 te1_b, 5 ge1_W, 6 ge1_b,
  // mix_a[i] at 7+5i: Wq,Wk,Wv,Wo,bo ; mix_b[i] at 17+5i,
  // 27 te2_W, 28 te2_b, 29 ge2_W, 30 ge2_b,
  // enc[l] at 31+12l: Win,bin,Wo,bo,g1,b1,W1,c1,W2,c2,g2,b2,
  // 55 ali_W, 56 ali_b, 57 rec_W, 58 rec_b
  auto F = [&](int i) { return (const float*)d_in[i]; };

  char* wsb = (char*)d_ws;
  size_t off = 0;
  auto alloc  = [&](size_t bytes) -> void* {
    void* p = wsb + off;
    off += (bytes + 255) & ~(size_t)255;
    return p;
  };
  auto allocF = [&](size_t n) { return (float*)alloc(n * sizeof(float)); };
  auto allocH = [&](size_t n) { return (bf16*)alloc(n * sizeof(bf16)); };

  auto cvtP = [&](const float* src, bf16* dst, size_t n) {
    int blocks = (int)((n + 2047) / 2048);
    if (blocks > 4096) blocks = 4096;
    k_f32_to_bf16<<<blocks, 256, 0, stream>>>(src, dst, (int)n);
  };
  auto cvt = [&](int idx, size_t n) -> bf16* {
    bf16* d = allocH(n);
    cvtP((const float*)d_in[idx], d, n);
    return d;
  };

  auto gemm = [&](const bf16* A, int lda, const bf16* Bm, int ldb,
                  const float* bias, float* C, bf16* Cb, int ldc,
                  int M, int N, int K, bool relu = false, bool transB = false) {
    dim3 gr(N / 128, M / 128);
    if (!transB) {
      if (!relu) k_gemm_wmma<false, false><<<gr, 256, 0, stream>>>(A, lda, Bm, ldb, bias, C, Cb, ldc, K);
      else       k_gemm_wmma<false, true ><<<gr, 256, 0, stream>>>(A, lda, Bm, ldb, bias, C, Cb, ldc, K);
    } else {
      if (!relu) k_gemm_wmma<true,  false><<<gr, 256, 0, stream>>>(A, lda, Bm, ldb, bias, C, Cb, ldc, K);
      else       k_gemm_wmma<true,  true ><<<gr, 256, 0, stream>>>(A, lda, Bm, ldb, bias, C, Cb, ldc, K);
    }
  };

  // ---- weights -> bf16 (mixer Wq/Wk skipped: dead) ----
  bf16* w_te1 = cvt(3, (size_t)1024 * 2048);
  bf16* w_ge1 = cvt(5, (size_t)1024 * 2048);
  bf16 *w_aWv[2], *w_aWo[2], *w_bWv[2], *w_bWo[2];
  for (int i = 0; i < 2; ++i) {
    w_aWv[i] = cvt(7 + 5 * i + 2, (size_t)1024 * 1024);
    w_aWo[i] = cvt(7 + 5 * i + 3, (size_t)1024 * 1024);
    w_bWv[i] = cvt(17 + 5 * i + 2, (size_t)1024 * 1024);
    w_bWo[i] = cvt(17 + 5 * i + 3, (size_t)1024 * 1024);
  }
  bf16* w_te2 = cvt(27, (size_t)1024 * 1024);
  bf16* w_ge2 = cvt(29, (size_t)1024 * 1024);
  bf16 *w_win[2], *w_wo[2], *w_w1[2], *w_w2[2];
  for (int l = 0; l < 2; ++l) {
    const int b = 31 + 12 * l;
    w_win[l] = cvt(b + 0, (size_t)6144 * 2048);
    w_wo[l]  = cvt(b + 2, (size_t)2048 * 2048);
    w_w1[l]  = cvt(b + 6, (size_t)2048 * 2048);
    w_w2[l]  = cvt(b + 8, (size_t)2048 * 2048);
  }
  bf16* w_rec = cvt(57, (size_t)2048 * 2048);

  // ---- activations: f32 primaries + bf16 shadows (GEMM A operands) ----
  float* t   = allocF((size_t)2048 * 1024);  bf16* th   = allocH((size_t)2048 * 1024);
  float* g   = allocF((size_t)2048 * 1024);  bf16* gh   = allocH((size_t)2048 * 1024);
  float* tn  = allocF((size_t)2048 * 1024);  bf16* tnh  = allocH((size_t)2048 * 1024);
  float* gn  = allocF((size_t)2048 * 1024);  bf16* gnh  = allocH((size_t)2048 * 1024);
  float* tmp = allocF((size_t)2048 * 1024);  bf16* tmph = allocH((size_t)2048 * 1024);
  float* x   = allocF((size_t)2048 * 2048);  bf16* xh   = allocH((size_t)2048 * 2048);
  float* xb  = allocF((size_t)2048 * 2048);  bf16* xbh  = allocH((size_t)2048 * 2048);
  float* qkv = allocF((size_t)2048 * 6144);  bf16* qkvh = allocH((size_t)2048 * 6144);
  float* sc  = allocF((size_t)2048 * 2048);  bf16* sch  = allocH((size_t)2048 * 2048);
  float* ob  = allocF((size_t)2048 * 2048);  bf16* obh  = allocH((size_t)2048 * 2048);
  float* ff  = allocF((size_t)2048 * 2048);  bf16* ffh  = allocH((size_t)2048 * 2048);
  float* h1  = allocF((size_t)2048 * 2048);
  bf16* text_h  = allocH((size_t)2048 * 2048);
  bf16* graph_h = allocH((size_t)2048 * 2048);

  cvtP(F(0), text_h,  (size_t)2048 * 2048);
  cvtP(F(1), graph_h, (size_t)2048 * 2048);

  // ---- embeddings ----
  gemm(text_h,  2048, w_te1, 2048, F(4), t, th, 1024, 2048, 1024, 2048);
  gemm(graph_h, 2048, w_ge1, 2048, F(6), g, gh, 1024, 2048, 1024, 2048);

  // ---- mixer layers (cross-attn collapses to Wv->Wo chain) ----
  for (int i = 0; i < 2; ++i) {
    gemm(gh,   1024, w_aWv[i], 1024, nullptr,           tmp, tmph, 1024, 2048, 1024, 1024);
    gemm(tmph, 1024, w_aWo[i], 1024, F(7 + 5 * i + 4),  tn,  tnh,  1024, 2048, 1024, 1024);
    gemm(th,   1024, w_bWv[i], 1024, nullptr,           tmp, tmph, 1024, 2048, 1024, 1024);
    gemm(tmph, 1024, w_bWo[i], 1024, F(17 + 5 * i + 4), gn,  gnh,  1024, 2048, 1024, 1024);
    float* sf; bf16* sh;
    sf = t; t = tn; tn = sf;   sh = th; th = tnh; tnh = sh;
    sf = g; g = gn; gn = sf;   sh = gh; gh = gnh; gnh = sh;
  }

  // ---- projections, concat into x ----
  gemm(th, 1024, w_te2, 1024, F(28), x,        xh,        2048, 2048, 1024, 1024);
  gemm(gh, 1024, w_ge2, 1024, F(30), x + 1024, xh + 1024, 2048, 2048, 1024, 1024);

  // ---- encoder layers: self-attention over S=2048, 8 heads, d=256 ----
  for (int l = 0; l < 2; ++l) {
    const int bi = 31 + 12 * l;
    gemm(xh, 2048, w_win[l], 2048, F(bi + 1), qkv, qkvh, 6144, 2048, 6144, 2048);
    for (int h = 0; h < 8; ++h) {
      // scores_h = Q_h @ K_h^T  (NT: both row-major, K contiguous)
      gemm(qkvh + h * 256, 6144, qkvh + 2048 + h * 256, 6144, nullptr,
           sc, nullptr, 2048, 2048, 2048, 256);
      k_softmax_rows<<<2048, 256, 0, stream>>>(sc, sch, 2048, 0.0625f); // 1/sqrt(256)
      // attn_h = P @ V_h  (NN via LDS transpose of B tile)
      gemm(sch, 2048, qkvh + 4096 + h * 256, 6144, nullptr,
           ob + h * 256, obh + h * 256, 2048, 2048, 256, 2048, false, true);
    }
    gemm(obh, 2048, w_wo[l], 2048, F(bi + 3), h1, nullptr, 2048, 2048, 2048, 2048);
    k_add_ln<<<2048, 256, 0, stream>>>(x, h1, F(bi + 4), F(bi + 5), xb, xbh, 2048);
    gemm(xbh, 2048, w_w1[l], 2048, F(bi + 7), ff, ffh, 2048, 2048, 2048, 2048, true);
    gemm(ffh, 2048, w_w2[l], 2048, F(bi + 9), h1, nullptr, 2048, 2048, 2048, 2048);
    k_add_ln<<<2048, 256, 0, stream>>>(xb, h1, F(bi + 10), F(bi + 11), x, xh, 2048);
  }

  // ---- heads ----
  float* out0   = (float*)d_out;                 // output (2048 x 2048)
  float* outAli = out0 + (size_t)2048 * 2048;    // ali    (2048 x 2)
  float* outRec = outAli + (size_t)2048 * 2;     // rec    (2048 x 2048)

  (void)hipMemcpyAsync(out0, x, (size_t)2048 * 2048 * sizeof(float),
                       hipMemcpyDeviceToDevice, stream);
  k_ali<<<2048, 256, 0, stream>>>(x, F(55), F(56), outAli);
  gemm(xh, 2048, w_rec, 2048, F(58), outRec, nullptr, 2048, 2048, 2048, 2048);
  k_log_softmax_rows<<<2048, 256, 0, stream>>>(outRec, 2048);
}
